// SelfAttention_67044439490787
// MI455X (gfx1250) — compile-verified, matrix operand order checked by reading, not run on hardware
//
#include <hip/hip_runtime.h>
#include <hip/hip_bf16.h>

// ---------------------------------------------------------------------------
// Self-attention, B=4, T=4096, C=1024, fp32 in/out.  (MI455X / gfx1250, wave32)
//   * all GEMMs via v_wmma_f32_16x16x32_bf16 (bf16 in, fp32 accum)
//   * staging via GLOBAL_LOAD_ASYNC_TO_LDS_B128 (ASYNCcnt), no reg staging
//   * softmax fp32 in-register, scores bf16 (128MB, ~L2-resident)
//   * 1/sqrt(C) folded into Q
// ---------------------------------------------------------------------------

typedef __attribute__((ext_vector_type(16))) __bf16 v16bf;
typedef __attribute__((ext_vector_type(8)))  __bf16 v8bf;
typedef __attribute__((ext_vector_type(4)))  __bf16 v4bf;
typedef __attribute__((ext_vector_type(8)))  float  v8f;

#define TILE_M 128
#define TILE_N 256
#define TILE_K 32
#define LDS_STRIDE 40   // 32 + 8 bf16 pad -> 80B row pitch, conflict-free b128 reads

// ---- helpers ---------------------------------------------------------------

__device__ __forceinline__ float bf_lo(unsigned u) {
    union { unsigned x; float f; } c; c.x = u << 16; return c.f;
}
__device__ __forceinline__ float bf_hi(unsigned u) {
    union { unsigned x; float f; } c; c.x = u & 0xffff0000u; return c.f;
}
__device__ __forceinline__ unsigned short f2bf_bits(float f) {
    union { __bf16 h; unsigned short u; } c; c.h = (__bf16)f; return c.u;
}

// one lane: copy 16B global -> 16B LDS, async (ASYNCcnt), no VGPR data
__device__ __forceinline__ void async_g2l_b128(unsigned ldsByteOff, const void* gaddr) {
    asm volatile("global_load_async_to_lds_b128 %0, %1, off"
                 :: "v"(ldsByteOff), "v"(gaddr) : "memory");
}
__device__ __forceinline__ void wait_async0() {
    asm volatile("s_wait_asynccnt 0x0" ::: "memory");
}

// Fragment for v_wmma_f32_16x16x32_bf16 (CDNA5 ISA 16-bit A 16x32 layout):
//   lanes 0-15 : row = lane,    K = {0..7, 16..23}
//   lanes 16-31: row = lane-16, K = {8..15, 24..31}
__device__ __forceinline__ v16bf load_frag(const __bf16* p) {
    v8bf lo = *reinterpret_cast<const v8bf*>(p);
    v8bf hi = *reinterpret_cast<const v8bf*>(p + 16);
    return __builtin_shufflevector(lo, hi, 0,1,2,3,4,5,6,7,8,9,10,11,12,13,14,15);
}

// ---- conversion kernels ----------------------------------------------------

__global__ __launch_bounds__(256) void cvt_f32_to_bf16(
        const float* __restrict__ in, __bf16* __restrict__ out, long n) {
    long i = ((long)blockIdx.x * blockDim.x + threadIdx.x) * 4;
    if (i + 3 < n) {
        float4 f = *reinterpret_cast<const float4*>(in + i);
        v4bf o = { (__bf16)f.x, (__bf16)f.y, (__bf16)f.z, (__bf16)f.w };
        *reinterpret_cast<v4bf*>(out + i) = o;
    }
}

// Wt[n][k] = (bf16) W[k][n].  grid = (Nd/32, Kd/32), block = 256 (32x8).
__global__ __launch_bounds__(256) void transpose_cvt(
        const float* __restrict__ W, __bf16* __restrict__ Wt, int Kd, int Nd) {
    __shared__ float tile[32][33];
    int tx = threadIdx.x & 31, ty = threadIdx.x >> 5;
    int n0 = blockIdx.x * 32, k0 = blockIdx.y * 32;
#pragma unroll
    for (int s = 0; s < 4; ++s)
        tile[ty + 8 * s][tx] = W[(size_t)(k0 + ty + 8 * s) * Nd + n0 + tx];
    __syncthreads();
#pragma unroll
    for (int s = 0; s < 4; ++s)
        Wt[(size_t)(n0 + ty + 8 * s) * Kd + k0 + tx] = (__bf16)tile[tx][ty + 8 * s];
}

// ---- generic bf16 WMMA GEMM ------------------------------------------------
// Out[m][n] = epilogue( sum_k A[m][k] * Bt[n][k] )
// A: MxK row-major bf16 (lda), Bt: NxK row-major bf16 (ldb).
// Requires M%128==0, N%256==0, K%32==0 (true for all calls here).
// OUT_MODE: 0 = fp32 row-major, 1 = bf16 row-major,
//           2 = bf16 transposed per batch-of-trT-rows: Out[(m/trT)][n][m%trT]
// Macro tile 128x256, 8 waves (2x4), each wave 64x64 = 4x4 WMMA accums.
template <int OUT_MODE, bool HAS_BIAS>
__global__ __launch_bounds__(256) void gemm_bf16_wmma(
        const __bf16* __restrict__ A, size_t strideA, int lda,
        const __bf16* __restrict__ Bt, size_t strideB, int ldb,
        void* __restrict__ Out, size_t strideO, int ldo,
        const float* __restrict__ bias,
        int N, int K, float scale, int trT) {
    __shared__ __bf16 lA[2][TILE_M * LDS_STRIDE];   // 2 x 10 KB
    __shared__ __bf16 lB[2][TILE_N * LDS_STRIDE];   // 2 x 20 KB

    const int z = blockIdx.z;
    A  += (size_t)z * strideA;
    Bt += (size_t)z * strideB;

    const int t    = threadIdx.x;
    const int lane = t & 31;
    const int wv   = t >> 5;          // 0..7
    const int wm   = wv >> 2;         // 0..1  -> 64-row half
    const int wn   = wv & 3;          // 0..3  -> 64-col quarter
    const int fr   = lane & 15;
    const int kb   = (lane >> 4) * 8; // K-half select per ISA layout

    const int mBase = blockIdx.y * TILE_M;
    const int nBase = blockIdx.x * TILE_N;

    // ---- async staging setup: 4 threads/row x 16B; A: 2 rounds, B: 4 rounds
    const int ldRow = t >> 2;           // 0..63
    const int ldK   = (t & 3) * 8;      // bf16 offset {0,8,16,24}
    const __bf16* gA0 = A  + (size_t)(mBase + ldRow) * lda + ldK;
    const __bf16* gA1 = gA0 + (size_t)64 * lda;
    const __bf16* gB0 = Bt + (size_t)(nBase + ldRow) * ldb + ldK;
    const __bf16* gB1 = gB0 + (size_t)64  * ldb;
    const __bf16* gB2 = gB0 + (size_t)128 * ldb;
    const __bf16* gB3 = gB0 + (size_t)192 * ldb;

    const unsigned rowOffB = (unsigned)(ldRow * LDS_STRIDE + ldK) * 2u;
    const unsigned lAbase  = (unsigned)(size_t)&lA[0][0];
    const unsigned lBbase  = (unsigned)(size_t)&lB[0][0];
    const unsigned bufA    = (unsigned)(TILE_M * LDS_STRIDE * 2);  // bytes/buffer
    const unsigned bufB    = (unsigned)(TILE_N * LDS_STRIDE * 2);
    const unsigned rstride = (unsigned)(64 * LDS_STRIDE * 2);      // 64 rows

    auto stage = [&](int buf, int kt) {
        const size_t ko = (size_t)kt * TILE_K;
        const unsigned a = lAbase + buf * bufA + rowOffB;
        const unsigned b = lBbase + buf * bufB + rowOffB;
        async_g2l_b128(a,               gA0 + ko);
        async_g2l_b128(a + rstride,     gA1 + ko);
        async_g2l_b128(b,               gB0 + ko);
        async_g2l_b128(b + rstride,     gB1 + ko);
        async_g2l_b128(b + 2 * rstride, gB2 + ko);
        async_g2l_b128(b + 3 * rstride, gB3 + ko);
    };

    v8f acc[4][4];
#pragma unroll
    for (int i = 0; i < 4; ++i)
#pragma unroll
        for (int j = 0; j < 4; ++j)
#pragma unroll
            for (int r = 0; r < 8; ++r) acc[i][j][r] = 0.0f;

    const int KT = K / TILE_K;

    stage(0, 0);
    wait_async0();
    __syncthreads();

    for (int kt = 0; kt < KT; ++kt) {
        const int  cur  = kt & 1;
        const bool more = (kt + 1 < KT);
        if (more) stage(cur ^ 1, kt + 1);   // DMA next tile while we compute

        v16bf af[4];
#pragma unroll
        for (int i = 0; i < 4; ++i)
            af[i] = load_frag(&lA[cur][(wm * 64 + i * 16 + fr) * LDS_STRIDE + kb]);
#pragma unroll
        for (int j = 0; j < 4; ++j) {
            v16bf bf = load_frag(&lB[cur][(wn * 64 + j * 16 + fr) * LDS_STRIDE + kb]);
#pragma unroll
            for (int i = 0; i < 4; ++i)
                acc[i][j] = __builtin_amdgcn_wmma_f32_16x16x32_bf16(
                    false, af[i], false, bf, (short)0, acc[i][j], false, false);
        }

        if (more) {
            wait_async0();      // our DMAs into nxt landed
            __syncthreads();    // everyone's landed; cur fully consumed
        }
    }

    // epilogue: C/D layout -> VGPR r: lanes 0-15 M=r, lanes 16-31 M=r+8; N=lane%16
    const int mW = mBase + wm * 64 + ((lane >> 4) << 3);
    const int nW = nBase + wn * 64 + (lane & 15);
#pragma unroll
    for (int i = 0; i < 4; ++i) {
#pragma unroll
        for (int j = 0; j < 4; ++j) {
            const int n = nW + j * 16;
            const float bsv = HAS_BIAS ? bias[n] : 0.0f;
#pragma unroll
            for (int r = 0; r < 8; ++r) {
                const int m = mW + i * 16 + r;
                const float v = (acc[i][j][r] + bsv) * scale;
                if (OUT_MODE == 0) {
                    float* o = (float*)Out + (size_t)z * strideO;
                    o[(size_t)m * ldo + n] = v;
                } else if (OUT_MODE == 1) {
                    __bf16* o = (__bf16*)Out + (size_t)z * strideO;
                    o[(size_t)m * ldo + n] = (__bf16)v;
                } else { // transposed (Vt): Out[b][n][tt]
                    const int b  = m / trT;
                    const int tt = m - b * trT;
                    ((__bf16*)Out)[((size_t)b * N + n) * trT + tt] = (__bf16)v;
                }
            }
        }
    }
}

// ---- in-place row softmax over bf16 rows of length 4096 --------------------
// One wave per row (row = 2048 dwords -> 16 uint4 per lane, all in registers).
__global__ __launch_bounds__(256) void softmax_rows4096_bf16(unsigned int* __restrict__ S) {
    const int lane = threadIdx.x & 31;
    const int wv   = threadIdx.x >> 5;
    const size_t row = (size_t)blockIdx.x * 8 + wv;
    uint4* rp = reinterpret_cast<uint4*>(S + row * 2048);

    uint4 d[16];
#pragma unroll
    for (int i = 0; i < 16; ++i) d[i] = rp[i * 32 + lane];

    float m = -1e30f;
#pragma unroll
    for (int i = 0; i < 16; ++i) {
        const unsigned* w = reinterpret_cast<const unsigned*>(&d[i]);
#pragma unroll
        for (int c = 0; c < 4; ++c)
            m = fmaxf(m, fmaxf(bf_lo(w[c]), bf_hi(w[c])));
    }
#pragma unroll
    for (int o = 16; o > 0; o >>= 1) m = fmaxf(m, __shfl_xor(m, o, 32));

    float s = 0.0f;
#pragma unroll
    for (int i = 0; i < 16; ++i) {
        unsigned* w = reinterpret_cast<unsigned*>(&d[i]);
#pragma unroll
        for (int c = 0; c < 4; ++c) {
            float a = __expf(bf_lo(w[c]) - m);
            float b = __expf(bf_hi(w[c]) - m);
            s += a + b;
            w[c] = (unsigned)f2bf_bits(a) | ((unsigned)f2bf_bits(b) << 16);
        }
    }
#pragma unroll
    for (int o = 16; o > 0; o >>= 1) s += __shfl_xor(s, o, 32);
    const float inv = 1.0f / s;

#pragma unroll
    for (int i = 0; i < 16; ++i) {
        unsigned* w = reinterpret_cast<unsigned*>(&d[i]);
#pragma unroll
        for (int c = 0; c < 4; ++c) {
            float a = bf_lo(w[c]) * inv;
            float b = bf_hi(w[c]) * inv;
            w[c] = (unsigned)f2bf_bits(a) | ((unsigned)f2bf_bits(b) << 16);
        }
        rp[i * 32 + lane] = d[i];
    }
}

// ---- host orchestration ----------------------------------------------------

extern "C" void kernel_launch(void* const* d_in, const int* in_sizes, int n_in,
                              void* d_out, int out_size, void* d_ws, size_t ws_size,
                              hipStream_t stream) {
    (void)in_sizes; (void)n_in; (void)out_size; (void)ws_size;
    const int B = 4, T = 4096, C = 1024;
    const long BT = (long)B * T;              // 16384

    const float* x  = (const float*)d_in[0];
    const float* Wq = (const float*)d_in[1];
    const float* bq = (const float*)d_in[2];
    const float* Wk = (const float*)d_in[3];
    const float* bk = (const float*)d_in[4];
    const float* Wv = (const float*)d_in[5];
    const float* bv = (const float*)d_in[6];
    const float* Wo = (const float*)d_in[7];
    const float* bo = (const float*)d_in[8];

    // workspace carve-up (256B aligned)
    char*  base = (char*)d_ws;
    size_t off  = 0;
    auto carve = [&](size_t bytes) {
        void* p = base + off;
        off += (bytes + 255) & ~(size_t)255;
        return p;
    };
    __bf16* xb  = (__bf16*)carve((size_t)BT * C * 2);        //  32 MB
    __bf16* WqT = (__bf16*)carve((size_t)C * C * 2);         //   2 MB
    __bf16* WkT = (__bf16*)carve((size_t)C * C * 2);
    __bf16* WvT = (__bf16*)carve((size_t)C * C * 2);
    __bf16* WoT = (__bf16*)carve((size_t)C * C * 2);
    __bf16* Q   = (__bf16*)carve((size_t)BT * C * 2);        //  32 MB (pre-scaled)
    __bf16* Kb  = (__bf16*)carve((size_t)BT * C * 2);        //  32 MB
    __bf16* Vt  = (__bf16*)carve((size_t)BT * C * 2);        //  32 MB  [B][C][T]
    __bf16* S   = (__bf16*)carve((size_t)B * T * T * 2);     // 128 MB  scores->probs
    __bf16* AO  = (__bf16*)carve((size_t)BT * C * 2);        //  32 MB

    // 1) fp32 -> bf16 conversions
    {
        long n = BT * C;
        cvt_f32_to_bf16<<<dim3((unsigned)(n / 4 / 256)), dim3(256), 0, stream>>>(x, xb, n);
        dim3 tg(C / 32, C / 32), tb(256);
        transpose_cvt<<<tg, tb, 0, stream>>>(Wq, WqT, C, C);
        transpose_cvt<<<tg, tb, 0, stream>>>(Wk, WkT, C, C);
        transpose_cvt<<<tg, tb, 0, stream>>>(Wv, WvT, C, C);
        transpose_cvt<<<tg, tb, 0, stream>>>(Wo, WoT, C, C);
    }

    const float invSqrtC = 1.0f / 32.0f;   // 1/sqrt(1024)
    dim3 blk(256);

    // 2) projections: Q (scaled) / K / Vt
    {
        dim3 g(C / TILE_N, BT / TILE_M, 1);
        gemm_bf16_wmma<1, true><<<g, blk, 0, stream>>>(xb, 0, C, WqT, 0, C, Q, 0, C,
                                                       bq, C, C, invSqrtC, 0);
        gemm_bf16_wmma<1, true><<<g, blk, 0, stream>>>(xb, 0, C, WkT, 0, C, Kb, 0, C,
                                                       bk, C, C, 1.0f, 0);
        gemm_bf16_wmma<2, true><<<g, blk, 0, stream>>>(xb, 0, C, WvT, 0, C, Vt, 0, C,
                                                       bv, C, C, 1.0f, T);
    }

    // 3) scores: S[b] = Q[b] * K[b]^T  (scale folded into Q)
    {
        dim3 g(T / TILE_N, T / TILE_M, B);
        gemm_bf16_wmma<1, false><<<g, blk, 0, stream>>>(Q, (size_t)T * C, C,
                                                        Kb, (size_t)T * C, C,
                                                        S, (size_t)T * T, T,
                                                        nullptr, T, C, 1.0f, 0);
    }

    // 4) softmax rows (in place, S becomes P)
    softmax_rows4096_bf16<<<dim3((unsigned)(BT / 8)), blk, 0, stream>>>((unsigned int*)S);

    // 5) AO[b] = P[b] * V[b]   (B operand = Vt rows = V columns)
    {
        dim3 g(C / TILE_N, T / TILE_M, B);
        gemm_bf16_wmma<1, false><<<g, blk, 0, stream>>>(S, (size_t)T * T, T,
                                                        Vt, (size_t)C * T, T,
                                                        AO, (size_t)T * C, C,
                                                        nullptr, C, T, 1.0f, 0);
    }

    // 6) out = AO * Wo + bo  -> fp32 d_out
    {
        dim3 g(C / TILE_N, BT / TILE_M, 1);
        gemm_bf16_wmma<0, true><<<g, blk, 0, stream>>>(AO, 0, C, WoT, 0, C,
                                                       d_out, 0, C,
                                                       bo, C, C, 1.0f, 0);
    }
}
